// PointTransformerLayer_83408264888943
// MI455X (gfx1250) — compile-verified, matrix operand order checked by reading, not run on hardware
//
#include <hip/hip_runtime.h>
#include <hip/hip_bf16.h>
#include <cfloat>
#include <cstdint>

// ---------------------------------------------------------------------------
// Point Transformer layer for MI455X (gfx1250, wave32, WMMA + TDM)
//   b=8, n=4096, din=dout=64, hpos=64, hattn=256, K=16
// Dominant math (34 GFLOP) runs on v_wmma_f32_16x16x32_f16.
// pos tiles for kNN staged into LDS via TENSOR_LOAD_TO_LDS (TDM).
// ---------------------------------------------------------------------------

typedef __attribute__((ext_vector_type(16))) _Float16 v16h;
typedef __attribute__((ext_vector_type(8)))  float    v8f;
typedef unsigned int u32x4 __attribute__((ext_vector_type(4)));
typedef int          i32x4 __attribute__((ext_vector_type(4)));
typedef int          i32x8 __attribute__((ext_vector_type(8)));

static constexpr int BB   = 8;     // batch
static constexpr int NN   = 4096;  // points
static constexpr int DD   = 64;    // feature dim
static constexpr int HH   = 256;   // hattn
static constexpr int KNB  = 16;    // neighbors
static constexpr long TOT = (long)BB * NN;   // 32768 points

// 16-bit A-matrix 16x32 element map (ISA 7.12.2):
//   element e (0..15) of lane L holds K index:
//   k = (e<8 ? e : e+8) + (L>=16 ? 8 : 0)
// i.e. two contiguous 8-element runs at +0 and +16 (per lane-half +8).
__device__ __forceinline__ int khi(int lane) { return (lane >= 16) ? 8 : 0; }

// ---------------------------------------------------------------------------
// TDM: 1D f32 tile copy global -> LDS via Tensor Data Mover.
// D# built per CDNA5 ISA §8 (group0: count/lds_addr/global_addr/type=2;
// group1: data_size=4B, tensor_dim0=tile_dim0=n, tensor_dim1=tile_dim1=1).
// Issued once per wave (EXEC ignored by TDM); waits TENSORcnt before return.
// ---------------------------------------------------------------------------
#if __has_builtin(__builtin_amdgcn_tensor_load_to_lds)
#define HAVE_TDM 1
__device__ __forceinline__ void tdm_load_f32_1d(unsigned lds_byte_addr,
                                                const float* gptr, unsigned n) {
  unsigned long long ga = (unsigned long long)(uintptr_t)gptr;
  u32x4 g0;
  g0[0] = 1u;                                             // count = 1
  g0[1] = lds_byte_addr;                                  // lds_addr
  g0[2] = (unsigned)(ga & 0xFFFFFFFFu);                   // global_addr lo
  g0[3] = (unsigned)((ga >> 32) & 0x01FFFFFFu) | (2u << 30); // hi | type=2
  i32x8 g1;
  g1[0] = (int)(2u << 16);                                // data_size = 4B
  g1[1] = (int)((n & 0xFFFFu) << 16);                     // tensor_dim0[15:0]
  g1[2] = (int)(((n >> 16) & 0xFFFFu) | (1u << 16));      // dim0 hi | dim1=1
  g1[3] = (int)((n & 0xFFFFu) << 16);                     // dim1 hi=0 | tile_dim0
  g1[4] = 1;                                              // tile_dim1=1, dim2=0
  g1[5] = (int)n;                                         // dim0_stride lo
  g1[6] = 0;
  g1[7] = 0;
  i32x4 z4 = {0, 0, 0, 0};
#if defined(__clang_major__) && (__clang_major__ >= 23)
  i32x8 z8 = {0, 0, 0, 0, 0, 0, 0, 0};
  __builtin_amdgcn_tensor_load_to_lds(g0, g1, z4, z4, z8, 0);
#else
  __builtin_amdgcn_tensor_load_to_lds(g0, g1, z4, z4, 0);
#endif
  __builtin_amdgcn_s_wait_tensorcnt(0);
}
#else
#define HAVE_TDM 0
#endif

// ---------------------------------------------------------------------------
// Generic f32-in / f32-out GEMM, f16 WMMA compute.  C[M,N] = A[M,K]@B[K,N]+bias
// One wave per 16x16 output tile.  M,N,K multiples of 16/16/32.
// ---------------------------------------------------------------------------
__global__ void gemm_wmma_f16_kernel(const float* __restrict__ A,
                                     const float* __restrict__ B,
                                     const float* __restrict__ bias,
                                     float* __restrict__ C,
                                     int M, int N, int Kd) {
  const int lane = threadIdx.x & 31;
  const int wave = threadIdx.x >> 5;
  const int wpb  = blockDim.x >> 5;
  const int ntiles = N >> 4;
  const int mtiles = M >> 4;
  int tile = blockIdx.x * wpb + wave;
  if (tile >= mtiles * ntiles) return;          // wave-uniform exit (EXEC all-1)
  const int mt = tile / ntiles, nt = tile % ntiles;

  const int m   = lane & 15;
  const int row = mt * 16 + m;
  const int col = nt * 16 + m;                  // lane&15 doubles as N index
  const int hi  = khi(lane);

  float bv = bias ? bias[col] : 0.0f;
  v8f acc = {bv, bv, bv, bv, bv, bv, bv, bv};

  for (int k0 = 0; k0 < Kd; k0 += 32) {
    // A fragment: two contiguous 8-float runs -> global_load_b128 pairs
    v16h a;
    {
      const float* p = A + (long)row * Kd + k0 + hi;
#pragma unroll
      for (int e = 0; e < 8; ++e) a[e]     = (_Float16)p[e];
#pragma unroll
      for (int e = 0; e < 8; ++e) a[8 + e] = (_Float16)p[16 + e];
    }
    // B fragment: same k map, column-strided (B is K x N row-major)
    v16h b;
    {
      const float* p = B + (long)(k0 + hi) * N + col;
#pragma unroll
      for (int e = 0; e < 8; ++e) b[e]     = (_Float16)p[(long)e * N];
#pragma unroll
      for (int e = 0; e < 8; ++e) b[8 + e] = (_Float16)p[(long)(16 + e) * N];
    }
    acc = __builtin_amdgcn_wmma_f32_16x16x32_f16(false, a, false, b,
                                                 (short)0, acc, false, false);
  }
  // D layout: vgpr r -> row (mt*16 + r + hi), column = col
  const int mbase = mt * 16 + hi;
#pragma unroll
  for (int r = 0; r < 8; ++r)
    C[(long)(mbase + r) * N + col] = acc[r];
}

// ---------------------------------------------------------------------------
// pe hidden layer: ph[p,h] = relu(pos[p,:3] @ w_pos1[:,h] + b_pos1[h])
// ---------------------------------------------------------------------------
__global__ void pe_hidden_kernel(const float* __restrict__ pos,
                                 const float* __restrict__ w1,
                                 const float* __restrict__ b1,
                                 float* __restrict__ ph) {
  long t = (long)blockIdx.x * blockDim.x + threadIdx.x;
  if (t >= TOT * 64) return;
  long p = t >> 6;
  int  h = (int)(t & 63);
  const float* pp = pos + p * 3;
  float acc = b1[h] + pp[0] * w1[0 * 64 + h] + pp[1] * w1[1 * 64 + h]
                    + pp[2] * w1[2 * 64 + h];
  ph[t] = fmaxf(acc, 0.0f);
}

// ---------------------------------------------------------------------------
// combine: s = q - k + pe ; vpe = v + pe   (qkv packed [p,192])
// ---------------------------------------------------------------------------
__global__ void combine_kernel(const float* __restrict__ qkv,
                               const float* __restrict__ pe,
                               float* __restrict__ s,
                               float* __restrict__ vpe) {
  long t = (long)blockIdx.x * blockDim.x + threadIdx.x;
  if (t >= TOT * 64) return;
  long p = t >> 6;
  int  c = (int)(t & 63);
  const float* row = qkv + p * 192;
  float pv = pe[t];
  s[t]   = row[c] - row[64 + c] + pv;
  vpe[t] = row[128 + c] + pv;
}

// ---------------------------------------------------------------------------
// kNN: per query, brute-force top-16 smallest squared distance.
// pos tiled through LDS (TDM async tile copy when available); top-16 kept as
// register insertion-sorted list.
// block = 256 threads = 256 queries; grid = (NN/256, BB)
// ---------------------------------------------------------------------------
static constexpr int TILE_J = 512;

__global__ void knn_kernel(const float* __restrict__ pos,
                           const unsigned char* __restrict__ mask,
                           int* __restrict__ idx) {
  __shared__ float         sp[TILE_J * 3];
  __shared__ unsigned char sm[TILE_J];

  const int b = blockIdx.y;
  const int i = blockIdx.x * 256 + threadIdx.x;
  const float* posb = pos + (long)b * NN * 3;

  const float px = posb[(long)i * 3 + 0];
  const float py = posb[(long)i * 3 + 1];
  const float pz = posb[(long)i * 3 + 2];
  const bool  mi = mask[(long)b * NN + i] != 0;

  float bd[KNB];
  int   bi[KNB];
#pragma unroll
  for (int r = 0; r < KNB; ++r) { bd[r] = FLT_MAX; bi[r] = 0; }

  for (int j0 = 0; j0 < NN; j0 += TILE_J) {
#if HAVE_TDM
    // One TDM descriptor per block-iteration, issued by wave 0 only
    // (TDM ignores EXEC; one instruction per wave). TENSORcnt waited inside.
    if (threadIdx.x < 32) {
      tdm_load_f32_1d((unsigned)(uintptr_t)sp, posb + (long)j0 * 3,
                      (unsigned)(TILE_J * 3));
    }
#else
    for (int t = threadIdx.x; t < TILE_J * 3; t += 256)
      sp[t] = posb[(long)j0 * 3 + t];
#endif
    for (int t = threadIdx.x; t < TILE_J; t += 256)
      sm[t] = mask[(long)b * NN + j0 + t];
    __syncthreads();

    for (int jj = 0; jj < TILE_J; ++jj) {
      float dx = px - sp[jj * 3 + 0];
      float dy = py - sp[jj * 3 + 1];
      float dz = pz - sp[jj * 3 + 2];
      float d  = dx * dx + dy * dy + dz * dz;
      if (!(mi && sm[jj])) d = FLT_MAX;
      if (d < bd[KNB - 1]) {
        bd[KNB - 1] = d;
        bi[KNB - 1] = j0 + jj;
#pragma unroll
        for (int r = KNB - 1; r > 0; --r) {
          if (bd[r] < bd[r - 1]) {
            float td = bd[r]; bd[r] = bd[r - 1]; bd[r - 1] = td;
            int   ti = bi[r]; bi[r] = bi[r - 1]; bi[r - 1] = ti;
          }
        }
      }
    }
    __syncthreads();
  }
  int* out = idx + ((long)b * NN + i) * KNB;
#pragma unroll
  for (int r = 0; r < KNB; ++r) out[r] = bi[r];
}

// ---------------------------------------------------------------------------
// Fused neighborhood attention. One wave per query.
//   A1 = s[idx,:]           (16 x 64)   f16 frags from global
//   h  = relu(A1 @ W1 + b1) (16 x 256)  WMMA, 32 tiles
//   sim= h @ W2 + b2        (16 x 64)   WMMA, 32 tiles (via LDS transpose)
//   softmax over neighbor axis, out = sum_k attn * vpe[idx,:]
// Weights pre-swizzled into LDS in exact B-fragment order (2x ds_load_b128
// per fragment).  Per-wave 1KB LDS scratch does D-layout -> A-layout.
// vpe gather rows prefetched (global_prefetch_b8) ~64 WMMAs ahead of use.
// ---------------------------------------------------------------------------
__global__ void attn_kernel(const float* __restrict__ s,
                            const float* __restrict__ vpe,
                            const int* __restrict__ idx,
                            const unsigned char* __restrict__ mask,
                            const float* __restrict__ w1,   // [64,256]
                            const float* __restrict__ b1,   // [256]
                            const float* __restrict__ w2,   // [256,64]
                            const float* __restrict__ b2,   // [64]
                            float* __restrict__ out) {
  extern __shared__ char smem[];
  _Float16* fw1 = (_Float16*)smem;          // 16 nt * 2 kc * 32 lane * 16 = 16384
  _Float16* fw2 = fw1 + 16384;              //  4 nt2* 8 kc * 32 lane * 16 = 16384
  _Float16* scr = fw2 + 16384;              //  8 waves * 16 x 32 halves  =  4096
  float*    sb1 = (float*)(scr + 8 * 512);  // 256
  float*    sb2 = sb1 + 256;                //  64

  // --- one-time weight swizzle into B-fragment order ------------------------
  for (int t = threadIdx.x; t < 16384; t += blockDim.x) {
    int e = t & 15, lane = (t >> 4) & 31, kc = (t >> 9) & 1, nt = t >> 10;
    int kk  = ((e < 8) ? e : e + 8) + khi(lane) + 32 * kc;
    int col = nt * 16 + (lane & 15);
    fw1[t] = (_Float16)w1[kk * HH + col];
  }
  for (int t = threadIdx.x; t < 16384; t += blockDim.x) {
    int e = t & 15, lane = (t >> 4) & 31, kc = (t >> 9) & 7, nt2 = t >> 12;
    int kk  = ((e < 8) ? e : e + 8) + khi(lane) + 32 * kc;
    int col = nt2 * 16 + (lane & 15);
    fw2[t] = (_Float16)w2[kk * DD + col];
  }
  for (int t = threadIdx.x; t < HH; t += blockDim.x) sb1[t] = b1[t];
  for (int t = threadIdx.x; t < DD; t += blockDim.x) sb2[t] = b2[t];
  __syncthreads();

  const int lane = threadIdx.x & 31;
  const int wid  = threadIdx.x >> 5;
  const int wpb  = blockDim.x >> 5;
  const int m    = lane & 15;      // neighbor row for A, N column for B/D
  const int hi   = khi(lane);
  _Float16* myscr = scr + wid * 512;

  for (long qi = (long)blockIdx.x * wpb + wid; qi < TOT;
       qi += (long)gridDim.x * wpb) {
    const int  b = (int)(qi / NN);
    const int  i = (int)(qi % NN);
    const int* idxq = idx + qi * KNB;

    // ---- neighbor ids for the softmax/gather phase, prefetch gather rows ---
    int jr[8];
#pragma unroll
    for (int r = 0; r < 8; ++r) {
      jr[r] = idxq[hi + r];
      __builtin_prefetch(vpe + ((long)b * NN + jr[r]) * DD, 0, 0);
    }

    // ---- A1 fragments: rows = s[b, idx[m], :] ------------------------------
    const int  jm    = idxq[m];
    const long sbase = ((long)b * NN + jm) * DD;
    v16h a1[2];
#pragma unroll
    for (int c = 0; c < 2; ++c) {
      const float* p = s + sbase + 32 * c + hi;
#pragma unroll
      for (int e = 0; e < 8; ++e) a1[c][e]     = (_Float16)p[e];
#pragma unroll
      for (int e = 0; e < 8; ++e) a1[c][8 + e] = (_Float16)p[16 + e];
    }

    // ---- sim accumulators (16 x 64), bias-initialised ----------------------
    v8f sim[4];
#pragma unroll
    for (int t2 = 0; t2 < 4; ++t2) {
      float bv = sb2[t2 * 16 + m];
      sim[t2] = (v8f){bv, bv, bv, bv, bv, bv, bv, bv};
    }

    // ---- fused GEMM1 (relu) -> LDS transpose -> GEMM2 ----------------------
#pragma unroll 4
    for (int kc = 0; kc < 8; ++kc) {          // 32-wide chunk of hattn
#pragma unroll
      for (int sub = 0; sub < 2; ++sub) {
        const int nt = kc * 2 + sub;
        float bv = sb1[nt * 16 + m];
        v8f h = {bv, bv, bv, bv, bv, bv, bv, bv};
        v16h bf0 = *(const v16h*)(fw1 + ((nt * 2 + 0) * 32 + lane) * 16);
        h = __builtin_amdgcn_wmma_f32_16x16x32_f16(false, a1[0], false, bf0,
                                                   (short)0, h, false, false);
        v16h bf1 = *(const v16h*)(fw1 + ((nt * 2 + 1) * 32 + lane) * 16);
        h = __builtin_amdgcn_wmma_f32_16x16x32_f16(false, a1[1], false, bf1,
                                                   (short)0, h, false, false);
        // relu + write D-layout tile into per-wave scratch (rows = neighbor m)
        const int colc  = sub * 16 + m;
        const int mbase = hi;
#pragma unroll
        for (int r = 0; r < 8; ++r)
          myscr[(mbase + r) * 32 + colc] = (_Float16)fmaxf(h[r], 0.0f);
      }
      // read back as 16-bit A fragment: two contiguous 16B runs per lane
      v16h a2;
      {
        const _Float16* p = myscr + m * 32 + hi;
#pragma unroll
        for (int e = 0; e < 8; ++e) a2[e]     = p[e];
#pragma unroll
        for (int e = 0; e < 8; ++e) a2[8 + e] = p[16 + e];
      }
#pragma unroll
      for (int t2 = 0; t2 < 4; ++t2) {
        v16h bf = *(const v16h*)(fw2 + ((t2 * 8 + kc) * 32 + lane) * 16);
        sim[t2] = __builtin_amdgcn_wmma_f32_16x16x32_f16(false, a2, false, bf,
                                                         (short)0, sim[t2],
                                                         false, false);
      }
    }

    // ---- mask, softmax over neighbors, weighted sum of vpe -----------------
    const bool mq = mask[(long)b * NN + i] != 0;
    float vld[8];
#pragma unroll
    for (int r = 0; r < 8; ++r)
      vld[r] = (mq && mask[(long)b * NN + jr[r]] != 0) ? 1.0f : 0.0f;

#pragma unroll
    for (int t2 = 0; t2 < 4; ++t2) {
      float vals[8];
#pragma unroll
      for (int r = 0; r < 8; ++r)
        vals[r] = (vld[r] > 0.0f) ? sim[t2][r] : -3.0e38f;
      float mx = vals[0];
#pragma unroll
      for (int r = 1; r < 8; ++r) mx = fmaxf(mx, vals[r]);
      mx = fmaxf(mx, __shfl_xor(mx, 16, 32));   // join m<8 / m>=8 halves
      float ex[8], se = 0.0f;
#pragma unroll
      for (int r = 0; r < 8; ++r) {
        ex[r] = (vld[r] > 0.0f) ? __expf(vals[r] - mx) : 0.0f;
        se += ex[r];
      }
      se += __shfl_xor(se, 16, 32);
      float inv = (se > 0.0f) ? (1.0f / se) : 0.0f;
      const int f = t2 * 16 + m;
      float acc = 0.0f;
#pragma unroll
      for (int r = 0; r < 8; ++r)
        acc += ex[r] * inv * vpe[((long)b * NN + jr[r]) * DD + f];
      acc += __shfl_xor(acc, 16, 32);
      if (lane < 16) out[qi * DD + f] = acc;
    }
  }
}

// ---------------------------------------------------------------------------
// host-side launch
// ---------------------------------------------------------------------------
extern "C" void kernel_launch(void* const* d_in, const int* in_sizes, int n_in,
                              void* d_out, int out_size, void* d_ws, size_t ws_size,
                              hipStream_t stream) {
  const float*         x      = (const float*)d_in[0];
  const float*         pos    = (const float*)d_in[1];
  const unsigned char* mask   = (const unsigned char*)d_in[2];
  const float*         w_qkv  = (const float*)d_in[3];
  const float*         w_pos1 = (const float*)d_in[4];
  const float*         b_pos1 = (const float*)d_in[5];
  const float*         w_pos2 = (const float*)d_in[6];
  const float*         b_pos2 = (const float*)d_in[7];
  const float*         w_att1 = (const float*)d_in[8];
  const float*         b_att1 = (const float*)d_in[9];
  const float*         w_att2 = (const float*)d_in[10];
  const float*         b_att2 = (const float*)d_in[11];
  float*               out    = (float*)d_out;

  // workspace partition (bytes)
  char* ws = (char*)d_ws;
  float* qkv = (float*)(ws);                       // 32768*192 f32 = 24 MB
  float* pe  = (float*)(ws + 25165824);            // 32768*64  f32 =  8 MB
  float* ph  = (float*)(ws + 33554432);            //  8 MB
  float* sA  = (float*)(ws + 41943040);            //  8 MB  (q - k + pe)
  float* vpe = (float*)(ws + 50331648);            //  8 MB  (v + pe)
  int*   idx = (int*)  (ws + 58720256);            //  2 MB

  // 1) pe hidden: relu(pos @ w_pos1 + b_pos1)
  {
    long total = TOT * 64;
    pe_hidden_kernel<<<(int)((total + 255) / 256), 256, 0, stream>>>(
        pos, w_pos1, b_pos1, ph);
  }
  // 2) pe = ph @ w_pos2 + b_pos2   (WMMA GEMM, M=32768,N=64,K=64)
  {
    int tiles = (32768 / 16) * (64 / 16);          // 8192 tiles, 8 waves/block
    gemm_wmma_f16_kernel<<<tiles / 8, 256, 0, stream>>>(
        ph, w_pos2, b_pos2, pe, 32768, 64, 64);
  }
  // 3) qkv = x @ w_qkv             (WMMA GEMM, M=32768,N=192,K=64)
  {
    int tiles = (32768 / 16) * (192 / 16);         // 24576 tiles
    gemm_wmma_f16_kernel<<<tiles / 8, 256, 0, stream>>>(
        x, w_qkv, nullptr, qkv, 32768, 192, 64);
  }
  // 4) s = q - k + pe ; vpe = v + pe
  {
    long total = TOT * 64;
    combine_kernel<<<(int)((total + 255) / 256), 256, 0, stream>>>(
        qkv, pe, sA, vpe);
  }
  // 5) kNN top-16 (TDM-staged pos tiles)
  {
    dim3 grid(NN / 256, BB);
    knn_kernel<<<grid, 256, 0, stream>>>(pos, mask, idx);
  }
  // 6) fused WMMA neighborhood attention
  {
    // LDS: 2*32KB fragged weights + 8KB scratch + biases
    size_t sh = 16384 * 2 * sizeof(_Float16)       // fw1+fw2
              + 8 * 512 * sizeof(_Float16)         // per-wave scratch
              + (256 + 64) * sizeof(float);        // biases
    attn_kernel<<<1024, 256, sh, stream>>>(
        sA, vpe, idx, mask, w_att1, b_att1, w_att2, b_att2, out);
  }
}
